// NeRFRenderer_75780402970652
// MI455X (gfx1250) — matrix operand order repaired and checked by reading, not compiled, not run on hardware
//
#include <hip/hip_runtime.h>
#include <hip/hip_bf16.h>
#include <math.h>

typedef __attribute__((ext_vector_type(16))) _Float16 v16h;
typedef __attribute__((ext_vector_type(8)))  float    v8f;

#define BOUND 2.0f
#define HDIM  64
#define ROWF  68          // padded row stride (floats) to avoid LDS bank conflicts
#define ROWH  136         // padded row stride in halves

__device__ __forceinline__ float sigmoidf_(float x){ return 1.0f/(1.0f+expf(-x)); }
__device__ __forceinline__ float softplusf_(float x){ return (x>20.0f)? x : log1pf(expf(x)); }
__device__ __forceinline__ float clamp01_(float x){ return fminf(fmaxf(x,0.0f),1.0f); }

// ---------------------------------------------------------------------------
// Kernel 1: per-ray AABB near/far + coarse z samples
// grid: N blocks of S threads
// ---------------------------------------------------------------------------
__global__ void k_setup(const float* __restrict__ ro, const float* __restrict__ rd,
                        float* __restrict__ nearv, float* __restrict__ farv,
                        float* __restrict__ zc, int N, int S)
{
    const int r = blockIdx.x;
    const int j = threadIdx.x;
    if (r >= N || j >= S) return;
    float nearm = -3.0e38f, farm = 3.0e38f;
    #pragma unroll
    for (int a=0;a<3;a++){
        const float o = ro[r*3+a];
        const float d = rd[r*3+a] + 1e-15f;
        const float t0 = (-BOUND - o)/d;
        const float t1 = ( BOUND - o)/d;
        nearm = fmaxf(nearm, fminf(t0,t1));
        farm  = fminf(farm,  fmaxf(t0,t1));
    }
    if (farm < nearm){ nearm = 1e9f; farm = 1e9f; }
    nearm = fmaxf(nearm, 0.05f);
    if (j == 0){ nearv[r] = nearm; farv[r] = farm; }
    const float t = (float)j/(float)(S-1);
    zc[r*S+j] = nearm + (farm-nearm)*t;
}

// ---------------------------------------------------------------------------
// Kernel 2: WMMA MLP field evaluation. One wave32 per 16-sample tile.
//   h1 = relu(pts@W1+b1)      4x v_wmma_f32_16x16x32_f16 (K padded 3->32)
//   h2 = relu(h1@W2+b2)       8x v_wmma (2 K-chunks x 4 N-tiles, chained C)
//   sigma = softplus(h2@Wsig); rgb = sigmoid([h2,dirs]@Wrgb + brgb)
// Host guarantees ntiles % (gridDim.x*8) == 0 so barriers are uniform.
// ---------------------------------------------------------------------------
__global__ __launch_bounds__(256) void k_field(
    const float* __restrict__ zv, const float* __restrict__ ro, const float* __restrict__ rd,
    const float* __restrict__ W1, const float* __restrict__ b1,
    const float* __restrict__ W2, const float* __restrict__ b2,
    const float* __restrict__ Wsig, const float* __restrict__ Wrgb, const float* __restrict__ brgb,
    float* __restrict__ sig_out, float* __restrict__ rgb_out, int ntiles, int S)
{
    __shared__ float sh_h[8][16*ROWF];           // per-wave h staging (f16 for h1, f32 for h2)
    __shared__ float sh_wsig[HDIM];
    __shared__ float sh_wrgb[(HDIM+3)*3];
    __shared__ float sh_brgb[3];

    const int tid = threadIdx.x;
    for (int i=tid;i<HDIM;i+=256)        sh_wsig[i] = Wsig[i];
    for (int i=tid;i<(HDIM+3)*3;i+=256)  sh_wrgb[i] = Wrgb[i];
    if (tid < 3) sh_brgb[tid] = brgb[tid];

    __builtin_prefetch(W1, 0, 0);                // global_prefetch_b8
    __builtin_prefetch(W2, 0, 0);

    const int  lane  = tid & 31;
    const int  wv    = tid >> 5;
    const int  nlane = lane & 15;
    const bool hiK   = lane >= 16;
    const int  kbase = hiK ? 16 : 0;             // B operand: lane-half selects K block of 16
    const int  mrow  = hiK ?  8 : 0;             // C/D layout: lanes 16-31 hold M=8..15

    // Wave-invariant B operands kept resident in VGPRs (f16)
    v16h B1r[4], B2r[2][4];
    float b1v[4], b2v[4];
    #pragma unroll
    for (int t=0;t<4;t++){
        const int n = 16*t + nlane;
        b1v[t] = b1[n]; b2v[t] = b2[n];
        #pragma unroll
        for (int h=0;h<16;h++){
            const int k = kbase + h;
            B1r[t][h] = (k < 3) ? (_Float16)W1[k*HDIM+n] : (_Float16)0.0f;
        }
        #pragma unroll
        for (int kc=0;kc<2;kc++){
            #pragma unroll
            for (int h=0;h<16;h++){
                const int k = 32*kc + kbase + h;
                B2r[kc][t][h] = (_Float16)W2[k*HDIM+n];
            }
        }
    }
    __syncthreads();

    float*    hbuf  = sh_h[wv];
    _Float16* hbufh = (_Float16*)hbuf;

    const int gwave  = blockIdx.x*8 + wv;
    const int nwaves = gridDim.x*8;
    const int iters  = ntiles / nwaves;          // exact by host construction

    for (int it=0; it<iters; ++it){
        const int tile = gwave + it*nwaves;
        __syncthreads();                          // previous head reads done

        const int   sidx = tile*16 + nlane;       // lanes 0-15 own distinct samples
        const int   ray  = sidx / S;
        const float zs   = zv[sidx];
        const float dx = rd[ray*3+0], dy = rd[ray*3+1], dz = rd[ray*3+2];
        const float px = fminf(fmaxf(ro[ray*3+0] + dx*zs, -BOUND), BOUND);
        const float py = fminf(fmaxf(ro[ray*3+1] + dy*zs, -BOUND), BOUND);
        const float pz = fminf(fmaxf(ro[ray*3+2] + dz*zs, -BOUND), BOUND);

        // A1: 16x32 f16, only K=0..2 nonzero (lanes>=16 hold K>=8 -> all zero)
        v16h a1;
        #pragma unroll
        for (int h=0;h<16;h++) a1[h] = (_Float16)0.0f;
        if (!hiK){ a1[0]=(_Float16)px; a1[1]=(_Float16)py; a1[2]=(_Float16)pz; }

        v8f cz;
        #pragma unroll
        for (int q=0;q<8;q++) cz[q] = 0.0f;

        // ----- layer 1: 4 WMMAs -----
        v8f c1[4];
        #pragma unroll
        for (int t=0;t<4;t++)
            c1[t] = __builtin_amdgcn_wmma_f32_16x16x32_f16(false, a1, false, B1r[t],
                                                           (short)0, cz, false, false);
        // bias + relu, stage h1 to LDS as f16 (D layout -> [M][N])
        #pragma unroll
        for (int t=0;t<4;t++){
            const int n = 16*t + nlane;
            #pragma unroll
            for (int r8=0;r8<8;r8++){
                const float v = fmaxf(c1[t][r8] + b1v[t], 0.0f);
                hbufh[(r8+mrow)*ROWH + n] = (_Float16)v;
            }
        }
        __syncthreads();

        // gather A2 chunks in A-operand layout (K split per lane-half)
        v16h a2[2];
        {
            const int koff = hiK ? 16 : 0;
            #pragma unroll
            for (int kc=0;kc<2;kc++){
                const _Float16* hp = &hbufh[nlane*ROWH + 32*kc + koff];
                #pragma unroll
                for (int h=0;h<8;h++){ a2[kc][h] = hp[h]; a2[kc][h+8] = hp[h+8]; }
            }
        }
        __syncthreads();

        // ----- layer 2: 8 WMMAs, accumulate across K chunks -----
        v8f c2[4];
        #pragma unroll
        for (int t=0;t<4;t++){
            c2[t] = __builtin_amdgcn_wmma_f32_16x16x32_f16(false, a2[0], false, B2r[0][t],
                                                           (short)0, cz, false, false);
            c2[t] = __builtin_amdgcn_wmma_f32_16x16x32_f16(false, a2[1], false, B2r[1][t],
                                                           (short)0, c2[t], false, false);
        }
        // bias + relu, stage h2 to LDS as f32
        #pragma unroll
        for (int t=0;t<4;t++){
            const int n = 16*t + nlane;
            #pragma unroll
            for (int r8=0;r8<8;r8++)
                hbuf[(r8+mrow)*ROWF + n] = fmaxf(c2[t][r8] + b2v[t], 0.0f);
        }
        __syncthreads();

        // ----- heads: lanes 0-15, one sample each -----
        if (!hiK){
            const float* hrow = &hbuf[nlane*ROWF];
            float ss = 0.0f, cr = sh_brgb[0], cg = sh_brgb[1], cb = sh_brgb[2];
            #pragma unroll 8
            for (int k=0;k<HDIM;k++){
                const float hv = hrow[k];
                ss += hv*sh_wsig[k];
                cr += hv*sh_wrgb[k*3+0];
                cg += hv*sh_wrgb[k*3+1];
                cb += hv*sh_wrgb[k*3+2];
            }
            cr += dx*sh_wrgb[64*3+0] + dy*sh_wrgb[65*3+0] + dz*sh_wrgb[66*3+0];
            cg += dx*sh_wrgb[64*3+1] + dy*sh_wrgb[65*3+1] + dz*sh_wrgb[66*3+1];
            cb += dx*sh_wrgb[64*3+2] + dy*sh_wrgb[65*3+2] + dz*sh_wrgb[66*3+2];
            sig_out[sidx]     = softplusf_(ss);
            rgb_out[sidx*3+0] = sigmoidf_(cr);
            rgb_out[sidx*3+1] = sigmoidf_(cg);
            rgb_out[sidx*3+2] = sigmoidf_(cb);
        }
    }
}

// ---------------------------------------------------------------------------
// Kernel 3: deterministic inverse-CDF importance resampling (one thread/ray)
// ---------------------------------------------------------------------------
__global__ void k_resample(const float* __restrict__ zc, const float* __restrict__ sigc,
                           const float* __restrict__ nearv, const float* __restrict__ farv,
                           float* __restrict__ zf, int N, int S, int U)
{
    const int r = blockIdx.x*blockDim.x + threadIdx.x;
    if (r >= N) return;
    const float* zr = &zc[(size_t)r*S];
    const float* sg = &sigc[(size_t)r*S];
    float z[128], w[128], cdf[128];
    for (int i=0;i<S;i++) z[i] = zr[i];
    const float nearm = nearv[r], farm = farv[r];
    const float sdist = (farm-nearm)/(float)S;

    float T = 1.0f;
    for (int i=0;i<S;i++){
        const float delta = (i < S-1) ? (z[i+1]-z[i]) : sdist;
        const float al = 1.0f - expf(-delta*sg[i]);
        w[i] = al*T;
        T *= (1.0f - al + 1e-15f);
    }
    // pdf over w[1..S-2], cdf length S-1 with cdf[0]=0
    float wsum = 0.0f;
    for (int j=1;j<=S-2;j++) wsum += w[j] + 1e-5f;
    cdf[0] = 0.0f;
    float acc = 0.0f;
    for (int j=1;j<=S-2;j++){ acc += (w[j]+1e-5f)/wsum; cdf[j] = acc; }

    float* out = &zf[(size_t)r*U];
    int ind = 1;                                  // cdf[0]=0 <= u always
    for (int k=0;k<U;k++){
        const float u = ((float)k + 0.5f)/(float)U;
        while (ind < S-1 && cdf[ind] <= u) ind++; // searchsorted right, monotone u
        const int below = ind - 1;
        const int above = (ind <= S-2) ? ind : S-2;
        const float cb = cdf[below], ca = cdf[above];
        const float bb = 0.5f*(z[below]+z[below+1]);   // z_mid bins
        const float ba = 0.5f*(z[above]+z[above+1]);
        float denom = ca - cb;
        if (denom < 1e-5f) denom = 1.0f;
        const float t = (u - cb)/denom;
        out[k] = bb + t*(ba - bb);
    }
}

// ---------------------------------------------------------------------------
// Kernel 4: merge (stable, both streams sorted) + composite (one thread/ray)
// ---------------------------------------------------------------------------
__global__ void k_composite(const float* __restrict__ zc, const float* __restrict__ sigc,
                            const float* __restrict__ rgbc,
                            const float* __restrict__ zf, const float* __restrict__ sigf,
                            const float* __restrict__ rgbf,
                            const float* __restrict__ nearv, const float* __restrict__ farv,
                            float* __restrict__ depth_out, float* __restrict__ img_out,
                            int N, int S, int U)
{
    const int r = blockIdx.x*blockDim.x + threadIdx.x;
    if (r >= N) return;
    const float* zcr = &zc  [(size_t)r*S];
    const float* scr = &sigc[(size_t)r*S];
    const float* rcr = &rgbc[(size_t)r*S*3];
    const float* zfr = &zf  [(size_t)r*U];
    const float* sfr = &sigf[(size_t)r*U];
    const float* rfr = &rgbf[(size_t)r*U*3];

    const float nearm = nearv[r], farm = farv[r];
    const float sdist = (farm-nearm)/(float)S;
    const float inv   = 1.0f/(farm-nearm);

    int i = 0, j = 0;
    float zci = zcr[0], zfj = zfr[0];
    float T = 1.0f, wsum = 0.0f, depth = 0.0f, c0 = 0.0f, c1 = 0.0f, c2 = 0.0f;
    float pzv = 0.0f, ps = 0.0f, pr0 = 0.0f, pr1 = 0.0f, pr2 = 0.0f;

    for (int k=0;k<S+U;k++){
        float czv, cs, q0, q1, q2;
        if (zci <= zfj){                          // ties -> coarse first (stable argsort)
            czv = zci; cs = scr[i]; q0 = rcr[3*i]; q1 = rcr[3*i+1]; q2 = rcr[3*i+2];
            i++; zci = (i < S) ? zcr[i] : 3.0e38f;
        } else {
            czv = zfj; cs = sfr[j]; q0 = rfr[3*j]; q1 = rfr[3*j+1]; q2 = rfr[3*j+2];
            j++; zfj = (j < U) ? zfr[j] : 3.0e38f;
        }
        if (k > 0){
            const float delta = czv - pzv;
            const float al  = 1.0f - expf(-delta*ps);
            const float wgt = al*T;
            T *= (1.0f - al + 1e-15f);
            wsum += wgt;
            depth += wgt*clamp01_((pzv - nearm)*inv);
            c0 += wgt*pr0; c1 += wgt*pr1; c2 += wgt*pr2;
        }
        pzv = czv; ps = cs; pr0 = q0; pr1 = q1; pr2 = q2;
    }
    {   // last sample uses sample_dist
        const float al  = 1.0f - expf(-sdist*ps);
        const float wgt = al*T;
        wsum += wgt;
        depth += wgt*clamp01_((pzv - nearm)*inv);
        c0 += wgt*pr0; c1 += wgt*pr1; c2 += wgt*pr2;
    }
    depth_out[r]   = depth;
    const float bg = 1.0f - wsum;                 // bg_color = 1
    img_out[3*r+0] = c0 + bg;
    img_out[3*r+1] = c1 + bg;
    img_out[3*r+2] = c2 + bg;
}

// ---------------------------------------------------------------------------
extern "C" void kernel_launch(void* const* d_in, const int* in_sizes, int n_in,
                              void* d_out, int out_size, void* d_ws, size_t ws_size,
                              hipStream_t stream)
{
    (void)n_in; (void)out_size; (void)ws_size;
    const float* rays_o = (const float*)d_in[0];
    const float* rays_d = (const float*)d_in[1];
    const float* W1     = (const float*)d_in[2];
    const float* b1     = (const float*)d_in[3];
    const float* W2     = (const float*)d_in[4];
    const float* b2     = (const float*)d_in[5];
    const float* Wsig   = (const float*)d_in[6];
    const float* Wrgb   = (const float*)d_in[7];
    const float* brgb   = (const float*)d_in[8];

    const int N = in_sizes[0]/3;   // 4096
    const int S = 128, U = 128;    // num_steps / upsample_steps (reference constants)

    float* ws = (float*)d_ws;
    size_t off = 0;
    float* nearv = ws + off; off += (size_t)N;
    float* farv  = ws + off; off += (size_t)N;
    float* z_c   = ws + off; off += (size_t)N*S;
    float* sig_c = ws + off; off += (size_t)N*S;
    float* rgb_c = ws + off; off += (size_t)N*S*3;
    float* z_f   = ws + off; off += (size_t)N*U;
    float* sig_f = ws + off; off += (size_t)N*U;
    float* rgb_f = ws + off; off += (size_t)N*U*3;

    float* out_depth = (float*)d_out;
    float* out_img   = out_depth + N;

    k_setup<<<N, S, 0, stream>>>(rays_o, rays_d, nearv, farv, z_c, N, S);

    // pick block count so ntiles % (blocks*8 waves) == 0 -> uniform barrier trips
    const int ntilesC = (N*S)/16;
    int blocksC = 1;
    for (int b=256;b>=1;b>>=1){ if (ntilesC % (b*8) == 0){ blocksC = b; break; } }
    k_field<<<blocksC, 256, 0, stream>>>(z_c, rays_o, rays_d, W1, b1, W2, b2,
                                         Wsig, Wrgb, brgb, sig_c, rgb_c, ntilesC, S);

    k_resample<<<(N+127)/128, 128, 0, stream>>>(z_c, sig_c, nearv, farv, z_f, N, S, U);

    const int ntilesF = (N*U)/16;
    int blocksF = 1;
    for (int b=256;b>=1;b>>=1){ if (ntilesF % (b*8) == 0){ blocksF = b; break; } }
    k_field<<<blocksF, 256, 0, stream>>>(z_f, rays_o, rays_d, W1, b1, W2, b2,
                                         Wsig, Wrgb, brgb, sig_f, rgb_f, ntilesF, U);

    k_composite<<<(N+127)/128, 128, 0, stream>>>(z_c, sig_c, rgb_c, z_f, sig_f, rgb_f,
                                                 nearv, farv, out_depth, out_img, N, S, U);
}
